// GAT_13778255086003
// MI455X (gfx1250) — matrix operand order repaired and checked by reading, not compiled, not run on hardware
//
#include <hip/hip_runtime.h>
#include <math.h>

// GAT fused kernels for gfx1250 (CDNA5, wave32, WMMA f16 16x16x32).
//
// N=4096, FIN=128, H=4, FOUT=64.
// Mask (64MB) is the only large operand -> memory-bound, ~3us floor at
// 23.3 TB/s. Flash-style single pass over columns; split-j (4 ways) for
// enough in-flight loads to saturate HBM, then a log-sum-exp combine.

#define GN   4096
#define GFIN 128
#define GH   4
#define GFO  64
#define GHF  (GH * GFO)   // 256
#define NSPLIT 4
#define JCHUNK (GN / NSPLIT)   // 1024 columns per split

typedef __attribute__((ext_vector_type(16))) _Float16 v16h;
typedef __attribute__((ext_vector_type(8)))  float    v8f;

union Frag16 {            // 32 bytes = one WMMA 16-bit A/B fragment per lane
    uint4 u[2];
    v16h  h;
};

__device__ __forceinline__ v8f wmma_f16(v16h a, v16h b, v8f c) {
    return __builtin_amdgcn_wmma_f32_16x16x32_f16(
        /*neg_a=*/false, a, /*neg_b=*/false, b,
        /*c_mod=*/(short)0, c, /*reuse_a=*/false, /*reuse_b=*/false);
}

// ---------------------------------------------------------------------------
// Kernel 0: f32 -> f16 conversions (+ transpose of skip_w to [FIN][H*FOUT])
// ---------------------------------------------------------------------------
__global__ void gat_convert(const float* __restrict__ x,
                            const float* __restrict__ proj,
                            const float* __restrict__ skw,
                            _Float16* __restrict__ x16,
                            _Float16* __restrict__ proj16,
                            _Float16* __restrict__ skwt16) {
    const int i  = blockIdx.x * blockDim.x + threadIdx.x;
    const int nx = GN * GFIN;          // 524288
    const int np = GH * GFIN * GFO;    // 32768
    const int ns = GFIN * GHF;         // 32768
    if (i < nx) x16[i] = (_Float16)x[i];
    if (i < np) proj16[i] = (_Float16)proj[i];
    if (i < ns) {
        const int k = i >> 8;          // / 256
        const int c = i & 255;
        skwt16[i] = (_Float16)skw[c * GFIN + k];   // transpose [256,128]->[128,256]
    }
}

// ---------------------------------------------------------------------------
// Kernel 1: h = x @ proj  (per head), skip = x @ skip_w^T, s_src/s_tgt dots.
// Block = 128 threads = 4 waves; wave w handles head w; block row-tile = 16.
// ---------------------------------------------------------------------------
__global__ __launch_bounds__(128)
void gat_proj(const _Float16* __restrict__ x16,
              const _Float16* __restrict__ proj16,
              const _Float16* __restrict__ skwt16,
              const float*    __restrict__ wsrc_g,
              const float*    __restrict__ wtgt_g,
              _Float16* __restrict__ h16,
              float*    __restrict__ s_src,
              float*    __restrict__ s_tgt,
              float*    __restrict__ skip_out) {
    const int lane = threadIdx.x & 31;
    const int hh   = threadIdx.x >> 5;     // head = wave id
    const int i0   = blockIdx.x * 16;
    const int m    = lane & 15;            // row within tile (A layout); col in B & C/D layouts
    const int hi   = lane >> 4;

    const v8f vzero = {0.f,0.f,0.f,0.f,0.f,0.f,0.f,0.f};
    v8f acc[4], accs[4];
#pragma unroll
    for (int nt = 0; nt < 4; ++nt) { acc[nt] = vzero; accs[nt] = vzero; }

#pragma unroll
    for (int k0 = 0; k0 < GFIN; k0 += 32) {
        // A fragment: lane l holds row m, K = {hi*8..hi*8+7, 16+hi*8..+7}
        Frag16 a;
        const _Float16* xr = x16 + (i0 + m) * GFIN + k0 + hi * 8;
        a.u[0] = *(const uint4*)(xr);
        a.u[1] = *(const uint4*)(xr + 16);

        // B fragments: lane l holds col m, K = {hi*8..hi*8+7, 16+hi*8..+7}
        const int kb = k0 + hi * 8;
        const _Float16* pb = proj16 + (hh * GFIN + kb) * GFO + m;
        const _Float16* sb = skwt16 + kb * GHF + hh * GFO + m;
#pragma unroll
        for (int nt = 0; nt < 4; ++nt) {
            Frag16 b, bs;
#pragma unroll
            for (int s = 0; s < 16; ++s) {
                const int kk = (s < 8) ? s : s + 8;
                b.h[s]  = pb[kk * GFO + nt * 16];
                bs.h[s] = sb[kk * GHF + nt * 16];
            }
            acc[nt]  = wmma_f16(a.h, b.h,  acc[nt]);
            accs[nt] = wmma_f16(a.h, bs.h, accs[nt]);
        }
    }

    // score vectors for my column slots
    float ws[4], wt[4];
#pragma unroll
    for (int nt = 0; nt < 4; ++nt) {
        ws[nt] = wsrc_g[hh * GFO + nt * 16 + m];
        wt[nt] = wtgt_g[hh * GFO + nt * 16 + m];
    }

    // C/D layout: VGPR r of lane l holds row (r + 8*hi), col (nt*16 + m)
#pragma unroll
    for (int r = 0; r < 8; ++r) {
        const int row = i0 + r + 8 * hi;
        float ps = 0.f, pt = 0.f;
#pragma unroll
        for (int nt = 0; nt < 4; ++nt) {
            const float v = acc[nt][r];
            h16[((size_t)hh * GN + row) * GFO + nt * 16 + m] = (_Float16)v;
            skip_out[(size_t)row * GHF + hh * GFO + nt * 16 + m] = accs[nt][r];
            ps += v * ws[nt];
            pt += v * wt[nt];
        }
        // reduce over the 16-lane half (col dimension)
#pragma unroll
        for (int d = 1; d < 16; d <<= 1) {
            ps += __shfl_xor(ps, d, 32);
            pt += __shfl_xor(pt, d, 32);
        }
        if (m == r) {                      // lane r writes row i0+r, lane r+16 writes i0+r+8
            s_src[hh * GN + row] = ps;
            s_tgt[hh * GN + row] = pt;
        }
    }
}

// ---------------------------------------------------------------------------
// Kernel 2: partial fused online-softmax attention over one j-chunk.
// grid = (N/16, NSPLIT); block = 128 threads = 4 waves (wave = head) so the
// mask tile is fetched once per (row-tile, split) and reused by all heads.
// Emits unnormalized O_s plus (m_s, l_s) per row for the combine step.
// ---------------------------------------------------------------------------
__global__ __launch_bounds__(128)
void gat_attn_part(const float*    __restrict__ mask,
                   const _Float16* __restrict__ h16,
                   const float*    __restrict__ s_src,
                   const float*    __restrict__ s_tgt,
                   float* __restrict__ part_O,   // [NSPLIT][H][N][64]
                   float* __restrict__ part_m,   // [NSPLIT][H][N]
                   float* __restrict__ part_l) { // [NSPLIT][H][N]
    const int lane = threadIdx.x & 31;
    const int hh   = threadIdx.x >> 5;
    const int i0   = blockIdx.x * 16;
    const int sp   = blockIdx.y;
    const int jbeg = sp * JCHUNK;
    const int jend = jbeg + JCHUNK;
    const int m    = lane & 15;
    const int hi   = lane >> 4;

    const float ssrc = s_src[hh * GN + i0 + m];         // stat row = m (both hi copies)
    const float* mrow = mask + (size_t)(i0 + m) * GN;
    const float* stg  = s_tgt + hh * GN;

    float mrun = -INFINITY, lrun = 0.f;
    const v8f vzero = {0.f,0.f,0.f,0.f,0.f,0.f,0.f,0.f};
    v8f acc[4];
#pragma unroll
    for (int nt = 0; nt < 4; ++nt) acc[nt] = vzero;

    for (int j0 = jbeg; j0 < jend; j0 += 32) {
        if (j0 + 96 < GN) __builtin_prefetch(mrow + j0 + 96, 0, 1);

        // my 16 columns: ja..ja+7 and ja+16..ja+23 (A-fragment K slots)
        const int ja = j0 + hi * 8;
        const float4 tq[4] = { *(const float4*)(stg + ja),
                               *(const float4*)(stg + ja + 4),
                               *(const float4*)(stg + ja + 16),
                               *(const float4*)(stg + ja + 20) };
        const float4 mq[4] = { *(const float4*)(mrow + ja),
                               *(const float4*)(mrow + ja + 4),
                               *(const float4*)(mrow + ja + 16),
                               *(const float4*)(mrow + ja + 20) };
        float p[16];
#pragma unroll
        for (int q = 0; q < 4; ++q) {
            const float* t  = (const float*)&tq[q];
            const float* mk = (const float*)&mq[q];
#pragma unroll
            for (int e = 0; e < 4; ++e) {
                float s = ssrc + t[e];
                s = (s > 0.f) ? s : 0.2f * s;       // leaky_relu(0.2)
                p[q * 4 + e] = s + mk[e];           // + mask
            }
        }

        // online softmax: tile max over my 16 cols, combine with pair lane
        float tmax = p[0];
#pragma unroll
        for (int i = 1; i < 16; ++i) tmax = fmaxf(tmax, p[i]);
        tmax = fmaxf(tmax, __shfl_xor(tmax, 16, 32));
        const float mnew  = fmaxf(mrun, tmax);
        const float scale = __expf(mrun - mnew);

        Frag16 a;
        float ssum = 0.f;
#pragma unroll
        for (int i = 0; i < 16; ++i) {
            const float e = __expf(p[i] - mnew);
            ssum += e;
            a.h[i] = (_Float16)e;                       // A fragment slot i
        }
        ssum += __shfl_xor(ssum, 16, 32);
        lrun = lrun * scale + ssum;
        mrun = mnew;

        // rescale accumulator rows (C/D row of VGPR r is r + 8*hi)
#pragma unroll
        for (int r = 0; r < 8; ++r) {
            const float f = __shfl(scale, r + 8 * hi, 32);
#pragma unroll
            for (int nt = 0; nt < 4; ++nt) acc[nt][r] = acc[nt][r] * f;
        }

        // V fragments: lane l holds col m, K rows {ja..ja+7, ja+16..ja+23} of h16[hh]
        const _Float16* vb = h16 + ((size_t)hh * GN + ja) * GFO + m;
        Frag16 b[4];
#pragma unroll
        for (int nt = 0; nt < 4; ++nt) {
#pragma unroll
            for (int s = 0; s < 16; ++s)
                b[nt].h[s] = vb[((s < 8) ? s : s + 8) * GFO + nt * 16];
        }
#pragma unroll
        for (int nt = 0; nt < 4; ++nt)
            acc[nt] = wmma_f16(a.h, b[nt].h, acc[nt]);
    }

    // emit partials (unnormalized O, per-row m and l)
    float* po = part_O + (((size_t)sp * GH + hh) * GN) * GFO;
#pragma unroll
    for (int r = 0; r < 8; ++r) {
        const int row = i0 + r + 8 * hi;
#pragma unroll
        for (int nt = 0; nt < 4; ++nt)
            po[(size_t)row * GFO + nt * 16 + m] = acc[nt][r];
    }
    if (hi == 0) {      // lane m owns stats of row i0+m
        const size_t si = ((size_t)sp * GH + hh) * GN + i0 + m;
        part_m[si] = mrun;
        part_l[si] = lrun;
    }
}

// ---------------------------------------------------------------------------
// Kernel 3: combine splits (log-sum-exp), + skip + bias, ELU, write output.
// One thread per output element (N x H*FOUT = 1M threads).
// ---------------------------------------------------------------------------
__global__ __launch_bounds__(256)
void gat_combine(const float* __restrict__ part_O,
                 const float* __restrict__ part_m,
                 const float* __restrict__ part_l,
                 const float* __restrict__ skip_out,
                 const float* __restrict__ bias,
                 float* __restrict__ out) {
    const int idx = blockIdx.x * blockDim.x + threadIdx.x;   // [0, N*GHF)
    const int row = idx >> 8;           // / 256
    const int c   = idx & 255;
    const int hh  = c >> 6;             // / 64
    const int col = c & 63;

    float ms[NSPLIT], ls[NSPLIT];
    float mstar = -INFINITY;
#pragma unroll
    for (int s = 0; s < NSPLIT; ++s) {
        const size_t si = ((size_t)s * GH + hh) * GN + row;
        ms[s] = part_m[si];
        ls[s] = part_l[si];
        mstar = fmaxf(mstar, ms[s]);
    }
    float lsum = 0.f, osum = 0.f;
#pragma unroll
    for (int s = 0; s < NSPLIT; ++s) {
        const float w = __expf(ms[s] - mstar);
        lsum += ls[s] * w;
        osum += part_O[(((size_t)s * GH + hh) * GN + row) * GFO + col] * w;
    }
    float v = osum / lsum + skip_out[(size_t)row * GHF + c] + bias[c];
    v = (v > 0.f) ? v : (__expf(v) - 1.0f);     // ELU(alpha=1)
    out[(size_t)row * GHF + c] = v;
}

// ---------------------------------------------------------------------------
extern "C" void kernel_launch(void* const* d_in, const int* in_sizes, int n_in,
                              void* d_out, int out_size, void* d_ws, size_t ws_size,
                              hipStream_t stream) {
    const float* x      = (const float*)d_in[0];   // [N,FIN]
    const float* mask   = (const float*)d_in[1];   // [N,N]
    const float* proj   = (const float*)d_in[2];   // [H,FIN,FOUT]
    const float* wsrc   = (const float*)d_in[3];   // [H,FOUT,1]
    const float* wtgt   = (const float*)d_in[4];   // [H,FOUT,1]
    const float* skw    = (const float*)d_in[5];   // [H*FOUT,FIN]
    const float* bias   = (const float*)d_in[6];   // [H*FOUT]
    float* out = (float*)d_out;                    // [N,H*FOUT]

    // workspace carve-out (~24.6 MB total)
    char* ws = (char*)d_ws;
    _Float16* x16    = (_Float16*)ws;  ws += (size_t)GN * GFIN * 2;         // 1 MB
    _Float16* proj16 = (_Float16*)ws;  ws += (size_t)GH * GFIN * GFO * 2;   // 64 KB
    _Float16* skwt16 = (_Float16*)ws;  ws += (size_t)GFIN * GHF * 2;        // 64 KB
    _Float16* h16    = (_Float16*)ws;  ws += (size_t)GH * GN * GFO * 2;     // 2 MB
    float*    ssrc   = (float*)ws;     ws += (size_t)GH * GN * 4;           // 64 KB
    float*    stgt   = (float*)ws;     ws += (size_t)GH * GN * 4;           // 64 KB
    float*    skip   = (float*)ws;     ws += (size_t)GN * GHF * 4;          // 4 MB
    float*    pO     = (float*)ws;     ws += (size_t)NSPLIT * GH * GN * GFO * 4; // 16 MB
    float*    pm     = (float*)ws;     ws += (size_t)NSPLIT * GH * GN * 4;  // 256 KB
    float*    pl     = (float*)ws;     ws += (size_t)NSPLIT * GH * GN * 4;  // 256 KB

    gat_convert<<<(GN * GFIN + 255) / 256, 256, 0, stream>>>(
        x, proj, skw, x16, proj16, skwt16);

    gat_proj<<<GN / 16, 128, 0, stream>>>(
        x16, proj16, skwt16, wsrc, wtgt, h16, ssrc, stgt, skip);

    gat_attn_part<<<dim3(GN / 16, NSPLIT), 128, 0, stream>>>(
        mask, h16, ssrc, stgt, pO, pm, pl);

    gat_combine<<<(GN * GHF + 255) / 256, 256, 0, stream>>>(
        pO, pm, pl, skip, bias, out);
}
